// PerPlaneMLP_46918222741858
// MI455X (gfx1250) — compile-verified, matrix operand order checked by reading, not run on hardware
//
#include <hip/hip_runtime.h>
#include <hip/hip_bf16.h>

// ---------------------------------------------------------------------------
// Per-plane MoE MLP for MI455X (gfx1250, wave32, WMMA bf16 16x16x32).
// ---------------------------------------------------------------------------

typedef __bf16 bf16_t;
typedef __attribute__((ext_vector_type(16))) __bf16 bf16x16;
typedef __attribute__((ext_vector_type(8)))  __bf16 bf16x8;
typedef __attribute__((ext_vector_type(8)))  float  f32x8;

#define L_PLANES 8
#define DIN      256
#define DHID     1024
#define DOUT     256
#define NWALK    32768
#define BM       32            // rows (walkers) per block
#define XLD      (DIN + 8)     // padded LDS leading dim for x tile (bf16 elems)
#define HLD      (DHID + 8)    // padded LDS leading dim for h tile (bf16 elems)

// ---------------------------------------------------------------------------
// Fragment loader: row-major 16xK bf16 tile -> WMMA A/B operand layout.
// 16-bit operand layout (ISA 7.12.2): lanes 0-15 hold row m = lane, K in
// {k0..k0+7, k0+16..k0+23}; lanes 16-31 hold row m = lane-16, K in
// {k0+8..k0+15, k0+24..k0+31}.  Both halves are two contiguous 16B loads.
// ---------------------------------------------------------------------------
__device__ __forceinline__ bf16x16 load_frag_rm(const bf16_t* base, int ld, int lane) {
    const int m   = lane & 15;
    const int sel = lane >> 4;          // 0 or 1
    const bf16_t* p = base + m * ld + sel * 8;
    bf16x8 lo = *(const bf16x8*)(p);
    bf16x8 hi = *(const bf16x8*)(p + 16);
    bf16x16 r;
#pragma unroll
    for (int i = 0; i < 8; ++i) { r[i] = lo[i]; r[i + 8] = hi[i]; }
    return r;
}

__device__ __forceinline__ f32x8 wmma_bf16(bf16x16 a, bf16x16 b, f32x8 c) {
    // (neg_a, A, neg_b, B, c_mod, C, reuse_a, reuse_b)
    return __builtin_amdgcn_wmma_f32_16x16x32_bf16(false, a, false, b,
                                                   (short)0, c, false, false);
}

__device__ __forceinline__ float gelu_exact(float v) {
    return 0.5f * v * (1.0f + erff(v * 0.70710678118654752440f));
}

// ---------------------------------------------------------------------------
// Kernel 1: bucket walkers by plane (gather lists in scratch).
// ---------------------------------------------------------------------------
__global__ __launch_bounds__(256)
void build_index_kernel(const int* __restrict__ plane_idx,
                        int* __restrict__ counts, int* __restrict__ slots) {
    int i = blockIdx.x * blockDim.x + threadIdx.x;
    if (i < NWALK) {
        int p   = plane_idx[i];
        int pos = atomicAdd(&counts[p], 1);
        slots[p * NWALK + pos] = i;
    }
}

// ---------------------------------------------------------------------------
// Kernel 2: one-time f32 -> bf16 conversion of both weight tensors.
// ---------------------------------------------------------------------------
__global__ __launch_bounds__(256)
void convert_weights_kernel(const float* __restrict__ W1, const float* __restrict__ W2,
                            bf16_t* __restrict__ W1b, bf16_t* __restrict__ W2b) {
    const int total = L_PLANES * DHID * DIN;   // == L_PLANES * DOUT * DHID
    for (int i = blockIdx.x * blockDim.x + threadIdx.x; i < total;
         i += gridDim.x * blockDim.x) {
        W1b[i] = (bf16_t)W1[i];
        W2b[i] = (bf16_t)W2[i];
    }
}

// ---------------------------------------------------------------------------
// Kernel 3: fused gather -> GEMM -> gelu -> GEMM -> scatter.
// Block: 256 threads = 8 wave32; 32 gathered rows of one plane.
// ---------------------------------------------------------------------------
__global__ __launch_bounds__(256)
void mlp_kernel(const float* __restrict__ x,
                const bf16_t* __restrict__ W1b, const float* __restrict__ b1,
                const bf16_t* __restrict__ W2b, const float* __restrict__ b2,
                const int* __restrict__ slots, const int* __restrict__ counts,
                float* __restrict__ out) {
    extern __shared__ char smem_raw[];
    bf16_t* xs = (bf16_t*)smem_raw;                                   // [BM][XLD]
    bf16_t* hs = (bf16_t*)(smem_raw + (size_t)BM * XLD * sizeof(bf16_t)); // [BM][HLD]

    const int p     = blockIdx.y;
    const int count = counts[p];
    const int row0  = blockIdx.x * BM;
    if (row0 >= count) return;          // uniform: EXEC stays all-ones for WMMA

    const int tid   = threadIdx.x;
    const int lane  = tid & 31;
    const int wv    = tid >> 5;
    const int nlane = lane & 15;
    const int selr  = (lane >> 4) * 8;

    const int*    psl = slots + p * NWALK;
    const bf16_t* W1p = W1b + (size_t)p * DHID * DIN;
    const bf16_t* W2p = W2b + (size_t)p * DOUT * DHID;
    const float*  b1p = b1 + p * DHID;
    const float*  b2p = b2 + p * DOUT;

    // ---- gather 32 rows of x, convert to bf16 in LDS ----------------------
    {
        const int r = tid >> 3;            // 0..31 : row within tile
        const int q = tid & 7;             // 8 threads per row, 32 cols each
        const int g = psl[min(row0 + r, count - 1)];
        const float4* src = (const float4*)(x + (size_t)g * DIN) + q * 8;
        bf16_t* dst = xs + r * XLD + q * 32;
#pragma unroll
        for (int c = 0; c < 4; ++c) {
            float4 v0 = src[c * 2 + 0];
            float4 v1 = src[c * 2 + 1];
            bf16x8 pk;
            pk[0] = (bf16_t)v0.x; pk[1] = (bf16_t)v0.y;
            pk[2] = (bf16_t)v0.z; pk[3] = (bf16_t)v0.w;
            pk[4] = (bf16_t)v1.x; pk[5] = (bf16_t)v1.y;
            pk[6] = (bf16_t)v1.z; pk[7] = (bf16_t)v1.w;
            *(bf16x8*)(dst + c * 8) = pk;
        }
    }
    __syncthreads();

    // ---- layer 1: h = gelu(x @ W1^T + b1) -> LDS bf16 ---------------------
    // Each wave owns 128 h-columns; split into two halves of 4 N-tiles to
    // bound accumulator VGPR pressure (64 VGPRs of acc per half).
#pragma unroll
    for (int half = 0; half < 2; ++half) {
        f32x8 acc[2][4];
#pragma unroll
        for (int mt = 0; mt < 2; ++mt)
#pragma unroll
            for (int nb = 0; nb < 4; ++nb) acc[mt][nb] = (f32x8)0.0f;

#pragma unroll
        for (int k0 = 0; k0 < DIN; k0 += 32) {
            bf16x16 a0 = load_frag_rm(xs + k0, XLD, lane);
            bf16x16 a1 = load_frag_rm(xs + 16 * XLD + k0, XLD, lane);
#pragma unroll
            for (int nb = 0; nb < 4; ++nb) {
                const int ncol = wv * 128 + half * 64 + nb * 16;
                bf16x16 bfr = load_frag_rm(W1p + (size_t)ncol * DIN + k0, DIN, lane);
                acc[0][nb] = wmma_bf16(a0, bfr, acc[0][nb]);
                acc[1][nb] = wmma_bf16(a1, bfr, acc[1][nb]);
            }
        }
        // epilogue: bias + exact gelu, store bf16 h tile
#pragma unroll
        for (int nb = 0; nb < 4; ++nb) {
            const int ncol = wv * 128 + half * 64 + nb * 16;
            const float bias = b1p[ncol + nlane];
#pragma unroll
            for (int mt = 0; mt < 2; ++mt) {
#pragma unroll
                for (int r = 0; r < 8; ++r) {
                    float v = acc[mt][nb][r] + bias;
                    hs[(mt * 16 + selr + r) * HLD + ncol + nlane] =
                        (bf16_t)gelu_exact(v);
                }
            }
        }
    }
    __syncthreads();

    // ---- layer 2: o = h @ W2^T + b2 ---------------------------------------
    f32x8 oacc[2][2];
#pragma unroll
    for (int mt = 0; mt < 2; ++mt)
#pragma unroll
        for (int nb = 0; nb < 2; ++nb) oacc[mt][nb] = (f32x8)0.0f;

#pragma unroll 4
    for (int k0 = 0; k0 < DHID; k0 += 32) {
        bf16x16 a0 = load_frag_rm(hs + k0, HLD, lane);
        bf16x16 a1 = load_frag_rm(hs + 16 * HLD + k0, HLD, lane);
#pragma unroll
        for (int nb = 0; nb < 2; ++nb) {
            const int ncol = wv * 32 + nb * 16;
            bf16x16 bfr = load_frag_rm(W2p + (size_t)ncol * DHID + k0, DHID, lane);
            oacc[0][nb] = wmma_bf16(a0, bfr, oacc[0][nb]);
            oacc[1][nb] = wmma_bf16(a1, bfr, oacc[1][nb]);
        }
    }

    // ---- scatter f32 rows back to out -------------------------------------
#pragma unroll
    for (int nb = 0; nb < 2; ++nb) {
        const int ncol = wv * 32 + nb * 16;
        const float bias = b2p[ncol + nlane];
#pragma unroll
        for (int mt = 0; mt < 2; ++mt) {
#pragma unroll
            for (int r = 0; r < 8; ++r) {
                const int lrow = mt * 16 + selr + r;
                if (row0 + lrow < count) {
                    const int g = psl[row0 + lrow];
                    out[(size_t)g * DOUT + ncol + nlane] = oacc[mt][nb][r] + bias;
                }
            }
        }
    }
}

// ---------------------------------------------------------------------------
// Host entry
// ---------------------------------------------------------------------------
extern "C" void kernel_launch(void* const* d_in, const int* in_sizes, int n_in,
                              void* d_out, int out_size, void* d_ws, size_t ws_size,
                              hipStream_t stream) {
    const float* x         = (const float*)d_in[0];
    const float* W1        = (const float*)d_in[1];
    const float* b1        = (const float*)d_in[2];
    const float* W2        = (const float*)d_in[3];
    const float* b2        = (const float*)d_in[4];
    const int*   plane_idx = (const int*)d_in[5];
    float*       out       = (float*)d_out;

    // scratch layout: counts | slots | W1(bf16) | W2(bf16)
    char* ws = (char*)d_ws;
    int*    counts = (int*)ws;
    int*    slots  = (int*)(ws + 256);
    bf16_t* W1b    = (bf16_t*)(ws + 256 + (size_t)L_PLANES * NWALK * sizeof(int));
    bf16_t* W2b    = W1b + (size_t)L_PLANES * DHID * DIN;

    hipMemsetAsync(counts, 0, L_PLANES * sizeof(int), stream);
    build_index_kernel<<<NWALK / 256, 256, 0, stream>>>(plane_idx, counts, slots);
    convert_weights_kernel<<<1024, 256, 0, stream>>>(W1, W2, W1b, W2b);

    dim3 grid(NWALK / BM, L_PLANES);
    size_t smem = (size_t)(BM * XLD + BM * HLD) * sizeof(bf16_t);  // ~83 KB of the 320 KB WGP LDS
    mlp_kernel<<<grid, 256, smem, stream>>>(x, W1b, b1, W2b, b2, slots, counts, out);
}
